// MultiHeadDotProductAttention_65996467470928
// MI455X (gfx1250) — compile-verified, matrix operand order checked by reading, not run on hardware
//
#include <hip/hip_runtime.h>
#include <hip/hip_bf16.h>

// ---------------------------------------------------------------------------
// MHA forward for MI455X (gfx1250, wave32, WMMA).
//   q = (Xq Wq^T + bq) * 0.125 ; k = Xkv Wk^T + bk ; v = Xkv Wv^T + bv
//   S = q k^T + mask ; P = softmax(S) ; O = P v ; out = O Wo^T + bo
// bf16 WMMA everywhere (f32 accumulate); flash-attention streaming softmax;
// K/V staged into LDS with CDNA5 async-to-LDS copies (double buffered);
// attention mask pre-packed to 1 bit/entry.
// ---------------------------------------------------------------------------

#define BS      2
#define SEQ     2048
#define EMB     1024
#define HEADS   16
#define HD      64
#define SCALE_F 0.125f
#define NEGF    (-1.0e10f)

#define KVB     64      // kv block staged in LDS
#define LDSROW  72      // padded LDS row stride (bf16 elems) = 144B (16B mult)

typedef __attribute__((ext_vector_type(16))) __bf16 v16bf;
typedef __attribute__((ext_vector_type(8)))  __bf16 v8bf;
typedef __attribute__((ext_vector_type(8)))  float  v8f;
typedef __attribute__((ext_vector_type(4)))  int    v4i;

#if defined(__gfx1250__) && __has_builtin(__builtin_amdgcn_global_load_async_to_lds_b128)
#define ASYNC_OK 1
#else
#define ASYNC_OK 0
#endif

__device__ __forceinline__ v8f wmma_bf16(v16bf a, v16bf b, v8f c) {
  // D = A(16x32 bf16) x B(32x16 bf16) + C(16x16 f32)
  return __builtin_amdgcn_wmma_f32_16x16x32_bf16(
      /*neg_a=*/false, a, /*neg_b=*/false, b,
      /*c_mod=*/(short)0, c, /*reuse_a=*/false, /*reuse_b=*/false);
}

// Load one 16x32 bf16 A/B fragment for this lane. p must already point at
// row_base + k0 + (lane>>4)*8.  Elements e<8 hold K = k0+8g+e, elements
// e>=8 hold K = k0+16+8g+(e-8)  (ISA 16-bit A-matrix layout, wave32).
__device__ __forceinline__ v16bf ldfrag(const __bf16* p) {
  v8bf lo = *(const v8bf*)(p);
  v8bf hi = *(const v8bf*)(p + 16);
  v16bf r;
#pragma unroll
  for (int e = 0; e < 8; ++e) { r[e] = lo[e]; r[e + 8] = hi[e]; }
  return r;
}

// 16-byte global -> LDS copy; async DMA path on gfx1250 (ASYNCcnt tracked).
// Builtin signature (from hipcc diagnostic): param0 = v4i* in addrspace(1).
__device__ __forceinline__ void cp16_g2l(const __bf16* g, __bf16* l) {
#if ASYNC_OK
  __builtin_amdgcn_global_load_async_to_lds_b128(
      (__attribute__((address_space(1))) v4i*)g,
      (__attribute__((address_space(3))) v4i*)l, /*offset=*/0, /*cpol=*/0);
#else
  *(v8bf*)l = *(const v8bf*)g;
#endif
}

__device__ __forceinline__ void stage_sync() {
#if ASYNC_OK
#if __has_builtin(__builtin_amdgcn_s_wait_asynccnt)
  __builtin_amdgcn_s_wait_asynccnt(0);
#else
  asm volatile("s_wait_asynccnt 0x0" ::: "memory");
#endif
#endif
  __syncthreads();
}

// ---------------------------------------------------------------------------
// fp32 -> bf16 conversion (grid-stride)
// ---------------------------------------------------------------------------
__global__ void cvt_f32_bf16_kernel(const float* __restrict__ s,
                                    __bf16* __restrict__ d, int n) {
  for (int i = blockIdx.x * blockDim.x + threadIdx.x; i < n;
       i += gridDim.x * blockDim.x)
    d[i] = (__bf16)s[i];
}

// ---------------------------------------------------------------------------
// Pack attn_mask (int32 > 0) into 1 bit per entry, 32 kv per word.
// ---------------------------------------------------------------------------
__global__ void pack_mask_kernel(const int* __restrict__ mask,
                                 unsigned* __restrict__ mbits, int nwords) {
  int w = blockIdx.x * blockDim.x + threadIdx.x;
  if (w >= nwords) return;
  const int* p = mask + (size_t)w * 32;
  unsigned bits = 0;
#pragma unroll
  for (int i = 0; i < 32; ++i) bits |= (p[i] > 0 ? 1u : 0u) << i;
  mbits[w] = bits;
}

// ---------------------------------------------------------------------------
// Fused QKV projection: C[m,n] = X[m,:] . W[n,:]  (m = b*SEQ+s, n = h*64+d)
// One wave computes a 64x64 tile.  blockIdx.z selects Q/K/V.
// Q stored [b,h,s,64] (pre-scaled), K stored [b,h,s,64], V stored [b,h,64,s].
// ---------------------------------------------------------------------------
__global__ __launch_bounds__(128) void qkv_proj_kernel(
    const __bf16* __restrict__ xq, const __bf16* __restrict__ xkv,
    const __bf16* __restrict__ Wq, const __bf16* __restrict__ Wk,
    const __bf16* __restrict__ Wv,
    const float* __restrict__ bq, const float* __restrict__ bk,
    const float* __restrict__ bv,
    __bf16* __restrict__ qh, __bf16* __restrict__ kh,
    __bf16* __restrict__ vT) {
  const int wave = threadIdx.x >> 5;
  const int lane = threadIdx.x & 31;
  const int g = lane >> 4;      // lane half
  const int c = lane & 15;      // row (A) / col (B) within tile
  const int proj = blockIdx.z;
  const int m0 = blockIdx.x * 64;
  const int n0 = (blockIdx.y * 4 + wave) * 64;

  const __bf16* X = (proj == 0) ? xq : xkv;
  const __bf16* W = (proj == 0) ? Wq : (proj == 1) ? Wk : Wv;
  const float* bias = (proj == 0) ? bq : (proj == 1) ? bk : bv;

  v8f acc[4][4];
  v8f zero = {};
#pragma unroll
  for (int i = 0; i < 4; ++i)
#pragma unroll
    for (int j = 0; j < 4; ++j) acc[i][j] = zero;

  for (int k0 = 0; k0 < EMB; k0 += 32) {
    v16bf a[4], bm[4];
#pragma unroll
    for (int i = 0; i < 4; ++i)
      a[i] = ldfrag(X + (size_t)(m0 + i * 16 + c) * EMB + k0 + g * 8);
#pragma unroll
    for (int j = 0; j < 4; ++j)
      bm[j] = ldfrag(W + (size_t)(n0 + j * 16 + c) * EMB + k0 + g * 8);
#pragma unroll
    for (int i = 0; i < 4; ++i)
#pragma unroll
      for (int j = 0; j < 4; ++j) acc[i][j] = wmma_bf16(a[i], bm[j], acc[i][j]);
  }

#pragma unroll
  for (int j = 0; j < 4; ++j) {
    const int n = n0 + j * 16 + c;       // C layout: lane -> column
    const float bn = bias[n];
    const int h = n >> 6, d = n & 63;
#pragma unroll
    for (int i = 0; i < 4; ++i) {
#pragma unroll
      for (int r = 0; r < 8; ++r) {
        const int m = m0 + i * 16 + r + 8 * g;  // C layout: vgpr -> row
        const int b = m >> 11, s = m & (SEQ - 1);
        const float val = acc[i][j][r] + bn;
        if (proj == 0)
          qh[(((size_t)(b * HEADS + h)) * SEQ + s) * HD + d] =
              (__bf16)(val * SCALE_F);
        else if (proj == 1)
          kh[(((size_t)(b * HEADS + h)) * SEQ + s) * HD + d] = (__bf16)val;
        else
          vT[(((size_t)(b * HEADS + h)) * HD + d) * SEQ + s] = (__bf16)val;
      }
    }
  }
}

// ---------------------------------------------------------------------------
// Flash attention.  Block = 4 waves, one (b,h), 128 query rows (32/wave).
// K/V staged to LDS in 64-kv blocks via async-to-LDS copies, double buffered.
// Computes S^T = K.Q^T so softmax-over-kv is a per-lane reduction (+ one
// shfl_xor 16), and P^T in C-layout feeds the PV WMMA B operand directly.
// O^T = V^T . P^T accumulated over kv blocks.
// ---------------------------------------------------------------------------
__global__ __launch_bounds__(128) void attn_kernel(
    const __bf16* __restrict__ qh, const __bf16* __restrict__ kh,
    const __bf16* __restrict__ vT, const unsigned* __restrict__ mbits,
    __bf16* __restrict__ attn_o) {
  __shared__ __bf16 lK[2][KVB * LDSROW];   // [kv][d]   rows, padded
  __shared__ __bf16 lV[2][KVB * LDSROW];   // [d][kv]   rows, padded

  const int tid = threadIdx.x;
  const int wave = tid >> 5;
  const int lane = tid & 31;
  const int g = lane >> 4;
  const int c = lane & 15;               // query column within tile
  const int bh = blockIdx.x;             // 0..31
  const int b = bh >> 4, h = bh & 15;
  const int q0 = blockIdx.y * 128 + wave * 32;

  const __bf16* Qb = qh + (size_t)bh * SEQ * HD;
  const __bf16* Kb = kh + (size_t)bh * SEQ * HD;
  const __bf16* Vb = vT + (size_t)bh * HD * SEQ;

  // cooperative K/V block stage: 64 rows x 128B each for K and V^T
  auto fill = [&](int bufi, int kv0) {
#pragma unroll
    for (int i = 0; i < 4; ++i) {
      const int ch = tid + 128 * i;          // 0..511
      const int row = ch >> 3;               // 0..63
      const int seg = (ch & 7) * 8;          // bf16 elems, 16B chunks
      cp16_g2l(Kb + (size_t)(kv0 + row) * HD + seg,
               &lK[bufi][row * LDSROW + seg]);
      cp16_g2l(Vb + (size_t)row * SEQ + kv0 + seg,
               &lV[bufi][row * LDSROW + seg]);
    }
  };

  // Q as B-fragments (lane -> query column), 2 q-tiles x 2 k-steps
  v16bf qf[2][2];
#pragma unroll
  for (int qt = 0; qt < 2; ++qt)
#pragma unroll
    for (int ks = 0; ks < 2; ++ks)
      qf[qt][ks] =
          ldfrag(Qb + (size_t)(q0 + qt * 16 + c) * HD + ks * 32 + g * 8);

  v8f zero = {};
  v8f o[2][4];
#pragma unroll
  for (int qt = 0; qt < 2; ++qt)
#pragma unroll
    for (int dt = 0; dt < 4; ++dt) o[qt][dt] = zero;
  float mrun[2] = {-3.0e30f, -3.0e30f};
  float lrun[2] = {0.0f, 0.0f};

  fill(0, 0);
  stage_sync();

  for (int blk = 0; blk < SEQ / KVB; ++blk) {
    const int cur = blk & 1;
    const int kv0 = blk * KVB;
    if (blk + 1 < SEQ / KVB) fill(cur ^ 1, kv0 + KVB);  // prefetch next

    const __bf16* Kl = lK[cur];
    const __bf16* Vl = lV[cur];

    // S^T block: 64(kv) x 32(q): 4 kv-tiles x 2 q-tiles, K-frags shared
    v8f st[2][4];
#pragma unroll
    for (int qt = 0; qt < 2; ++qt)
#pragma unroll
      for (int t = 0; t < 4; ++t) st[qt][t] = zero;
#pragma unroll
    for (int t = 0; t < 4; ++t) {
      const __bf16* kr = Kl + (t * 16 + c) * LDSROW;
      v16bf a0 = ldfrag(kr + g * 8);
      v16bf a1 = ldfrag(kr + 32 + g * 8);
#pragma unroll
      for (int qt = 0; qt < 2; ++qt) {
        st[qt][t] = wmma_bf16(a0, qf[qt][0], st[qt][t]);
        st[qt][t] = wmma_bf16(a1, qf[qt][1], st[qt][t]);
      }
    }

    // mask + online softmax (per lane over 32 kv values + one shfl)
    v16bf pf[2][2];
#pragma unroll
    for (int qt = 0; qt < 2; ++qt) {
      const int q = q0 + qt * 16 + c;
      const unsigned* mb = mbits + ((size_t)(b * SEQ + q) << 6) + (kv0 >> 5);
      const unsigned w0 = mb[0], w1 = mb[1];
#pragma unroll
      for (int t = 0; t < 4; ++t) {
        const unsigned byte =
            ((t < 2 ? w0 : w1) >> ((t & 1) * 16 + 8 * g)) & 0xffu;
#pragma unroll
        for (int r = 0; r < 8; ++r)
          st[qt][t][r] += ((byte >> r) & 1u) ? 0.0f : NEGF;
      }

      float tmax = st[qt][0][0];
#pragma unroll
      for (int t = 0; t < 4; ++t)
#pragma unroll
        for (int e = 0; e < 8; ++e) tmax = fmaxf(tmax, st[qt][t][e]);
      tmax = fmaxf(tmax, __shfl_xor(tmax, 16, 32));
      const float mnew = fmaxf(mrun[qt], tmax);
      const float alpha = __expf(mrun[qt] - mnew);

      float ladd = 0.0f;
#pragma unroll
      for (int e = 0; e < 16; ++e) {
        const float p0 = __expf(st[qt][e >> 3][e & 7] - mnew);
        const float p1 = __expf(st[qt][2 + (e >> 3)][e & 7] - mnew);
        ladd += p0 + p1;
        pf[qt][0][e] = (__bf16)p0;   // C-layout == B-fragment layout
        pf[qt][1][e] = (__bf16)p1;
      }
      ladd += __shfl_xor(ladd, 16, 32);
      lrun[qt] = lrun[qt] * alpha + ladd;
#pragma unroll
      for (int dt = 0; dt < 4; ++dt)
#pragma unroll
        for (int r = 0; r < 8; ++r) o[qt][dt][r] *= alpha;
      mrun[qt] = mnew;
    }

    // O^T += V^T(64 x 64kv) . P^T(64kv x 16q), V-frags shared across q-tiles
#pragma unroll
    for (int dt = 0; dt < 4; ++dt) {
      const __bf16* vr = Vl + (dt * 16 + c) * LDSROW;
      v16bf a0 = ldfrag(vr + g * 8);
      v16bf a1 = ldfrag(vr + 32 + g * 8);
#pragma unroll
      for (int qt = 0; qt < 2; ++qt) {
        o[qt][dt] = wmma_bf16(a0, pf[qt][0], o[qt][dt]);
        o[qt][dt] = wmma_bf16(a1, pf[qt][1], o[qt][dt]);
      }
    }

    stage_sync();  // own async loads done + all waves done reading `cur`
  }

#pragma unroll
  for (int qt = 0; qt < 2; ++qt) {
    const int q = q0 + qt * 16 + c;
    const float inv = 1.0f / lrun[qt];
#pragma unroll
    for (int dt = 0; dt < 4; ++dt) {
#pragma unroll
      for (int r = 0; r < 8; ++r) {
        const int d = dt * 16 + r + 8 * g;   // O^T rows = head dim
        attn_o[(((size_t)(b * SEQ + q)) * HEADS + h) * HD + d] =
            (__bf16)(o[qt][dt][r] * inv);
      }
    }
  }
}

// ---------------------------------------------------------------------------
// Output projection: out[m,n] = AO[m,:] . Wo[n,:] + bo[n]   (fp32 out)
// ---------------------------------------------------------------------------
__global__ __launch_bounds__(128) void oproj_kernel(
    const __bf16* __restrict__ ao, const __bf16* __restrict__ Wo,
    const float* __restrict__ bo, float* __restrict__ out) {
  const int wave = threadIdx.x >> 5;
  const int lane = threadIdx.x & 31;
  const int g = lane >> 4, c = lane & 15;
  const int m0 = blockIdx.x * 64;
  const int n0 = (blockIdx.y * 4 + wave) * 64;

  v8f acc[4][4];
  v8f zero = {};
#pragma unroll
  for (int i = 0; i < 4; ++i)
#pragma unroll
    for (int j = 0; j < 4; ++j) acc[i][j] = zero;

  for (int k0 = 0; k0 < EMB; k0 += 32) {
    v16bf a[4], bm[4];
#pragma unroll
    for (int i = 0; i < 4; ++i)
      a[i] = ldfrag(ao + (size_t)(m0 + i * 16 + c) * EMB + k0 + g * 8);
#pragma unroll
    for (int j = 0; j < 4; ++j)
      bm[j] = ldfrag(Wo + (size_t)(n0 + j * 16 + c) * EMB + k0 + g * 8);
#pragma unroll
    for (int i = 0; i < 4; ++i)
#pragma unroll
      for (int j = 0; j < 4; ++j) acc[i][j] = wmma_bf16(a[i], bm[j], acc[i][j]);
  }

#pragma unroll
  for (int j = 0; j < 4; ++j) {
    const int n = n0 + j * 16 + c;
    const float bn = bo[n];
#pragma unroll
    for (int i = 0; i < 4; ++i)
#pragma unroll
      for (int r = 0; r < 8; ++r) {
        const int m = m0 + i * 16 + r + 8 * g;
        out[(size_t)m * EMB + n] = acc[i][j][r] + bn;
      }
  }
}

// ---------------------------------------------------------------------------
// Launch
// ---------------------------------------------------------------------------
extern "C" void kernel_launch(void* const* d_in, const int* in_sizes, int n_in,
                              void* d_out, int out_size, void* d_ws,
                              size_t ws_size, hipStream_t stream) {
  const float* xq_f  = (const float*)d_in[0];
  const float* xkv_f = (const float*)d_in[1];
  const int*   mask  = (const int*)d_in[2];
  const float* Wq_f  = (const float*)d_in[3];
  const float* bq    = (const float*)d_in[4];
  const float* Wk_f  = (const float*)d_in[5];
  const float* bk    = (const float*)d_in[6];
  const float* Wv_f  = (const float*)d_in[7];
  const float* bv    = (const float*)d_in[8];
  const float* Wo_f  = (const float*)d_in[9];
  const float* bo    = (const float*)d_in[10];
  float* out = (float*)d_out;

  const int NX = BS * SEQ * EMB;            // 4,194,304
  const int NW = EMB * EMB;                 // 1,048,576
  const int NMASKW = BS * SEQ * SEQ / 32;   // 262,144 packed mask words

  __bf16* ws = (__bf16*)d_ws;               // ~57 MB scratch (element offsets)
  __bf16* xq_b  = ws;
  __bf16* xkv_b = xq_b + NX;
  __bf16* Wq_b  = xkv_b + NX;
  __bf16* Wk_b  = Wq_b + NW;
  __bf16* Wv_b  = Wk_b + NW;
  __bf16* Wo_b  = Wv_b + NW;
  __bf16* qh    = Wo_b + NW;                // [b,h,s,64], pre-scaled by 1/8
  __bf16* kh    = qh + NX;                  // [b,h,s,64]
  __bf16* vT    = kh + NX;                  // [b,h,64,s]
  __bf16* ao    = vT + NX;                  // [b,s,emb]
  unsigned* mbits = (unsigned*)(ao + NX);   // 1-bit mask, 1 MB

  // 1) fp32 -> bf16 converts + mask bit-pack
  cvt_f32_bf16_kernel<<<2048, 256, 0, stream>>>(xq_f, xq_b, NX);
  cvt_f32_bf16_kernel<<<2048, 256, 0, stream>>>(xkv_f, xkv_b, NX);
  cvt_f32_bf16_kernel<<<1024, 256, 0, stream>>>(Wq_f, Wq_b, NW);
  cvt_f32_bf16_kernel<<<1024, 256, 0, stream>>>(Wk_f, Wk_b, NW);
  cvt_f32_bf16_kernel<<<1024, 256, 0, stream>>>(Wv_f, Wv_b, NW);
  cvt_f32_bf16_kernel<<<1024, 256, 0, stream>>>(Wo_f, Wo_b, NW);
  pack_mask_kernel<<<(NMASKW + 255) / 256, 256, 0, stream>>>(mask, mbits,
                                                             NMASKW);

  // 2) QKV projections: M=4096 (64 tiles), N=1024 (16 tiles, 4 waves/block)
  qkv_proj_kernel<<<dim3(64, 4, 3), 128, 0, stream>>>(
      xq_b, xkv_b, Wq_b, Wk_b, Wv_b, bq, bk, bv, qh, kh, vT);

  // 3) flash attention: 32 (b,h) x 16 q-blocks of 128 rows (4 waves/block)
  attn_kernel<<<dim3(32, 16), 128, 0, stream>>>(qh, kh, vT, mbits, ao);

  // 4) output projection
  oproj_kernel<<<dim3(64, 4), 128, 0, stream>>>(ao, Wo_b, bo, out);
}
